// NeuralSSD_86672440033600
// MI455X (gfx1250) — matrix operand order repaired and checked by reading, not compiled, not run on hardware
//
#include <hip/hip_runtime.h>
#include <hip/hip_bf16.h>
#include <stdint.h>

typedef __attribute__((ext_vector_type(16))) __bf16 v16bf;
typedef __attribute__((ext_vector_type(8)))  __bf16 v8bf;
typedef __attribute__((ext_vector_type(8)))  float  v8f;
typedef __attribute__((ext_vector_type(4)))  unsigned int v4u;
typedef __attribute__((ext_vector_type(8)))  int    v8i;
typedef __attribute__((ext_vector_type(4)))  int    v4i;

#define B_Q    1024
#define C_K    262144
#define D_DIM  512
#define V_DIM  512
#define KTOP   8
#define CHUNK  4096
#define NCHUNK (C_K / CHUNK)   // 64 chunks along keys
#define MTILE  64              // queries per workgroup
#define NBLK   128             // keys per inner block
#define KC     64              // K-chunk staged in LDS

#define NEG_INF (-3.402823466e38f)

#if defined(__has_builtin)
#if __has_builtin(__builtin_amdgcn_tensor_load_to_lds) && __has_builtin(__builtin_amdgcn_s_wait_tensorcnt)
#define HAVE_TDM 1
#endif
#endif

// round-to-nearest-even f32 -> packed 2x bf16
__device__ inline unsigned pk_bf16(float a, float b) {
  unsigned ua = __float_as_uint(a), ub = __float_as_uint(b);
  ua += 0x7FFFu + ((ua >> 16) & 1u);
  ub += 0x7FFFu + ((ub >> 16) & 1u);
  return (ua >> 16) | (ub & 0xFFFF0000u);
}

// ---------------- prepass: normalize queries -> bf16 ----------------
__global__ __launch_bounds__(256) void prep_query(const float* __restrict__ q,
                                                  __bf16* __restrict__ qn) {
  int row  = blockIdx.x * 8 + (threadIdx.x >> 5);
  int lane = threadIdx.x & 31;
  const float4* src = (const float4*)(q + (size_t)row * D_DIM);
  float4 v[4];
  float s = 0.f;
#pragma unroll
  for (int i = 0; i < 4; ++i) {
    v[i] = src[lane + i * 32];
    s += v[i].x * v[i].x + v[i].y * v[i].y + v[i].z * v[i].z + v[i].w * v[i].w;
  }
#pragma unroll
  for (int m = 16; m >= 1; m >>= 1) s += __shfl_xor(s, m, 32);
  float inv = 1.0f / fmaxf(sqrtf(s), 1e-12f);
  uint2* dst = (uint2*)(qn + (size_t)row * D_DIM);
#pragma unroll
  for (int i = 0; i < 4; ++i) {
    float4 t = v[i];
    dst[lane + i * 32] = make_uint2(pk_bf16(t.x * inv, t.y * inv),
                                    pk_bf16(t.z * inv, t.w * inv));
  }
}

// ---------------- prepass: per-key inverse L2 norm ----------------
__global__ __launch_bounds__(256) void prep_keys(const float* __restrict__ k,
                                                 float* __restrict__ kinv) {
  int row  = blockIdx.x * 8 + (threadIdx.x >> 5);
  int lane = threadIdx.x & 31;
  const float4* src = (const float4*)(k + (size_t)row * D_DIM);
  float s = 0.f;
#pragma unroll
  for (int i = 0; i < 4; ++i) {
    float4 v = src[lane + i * 32];
    s += v.x * v.x + v.y * v.y + v.z * v.z + v.w * v.w;
  }
#pragma unroll
  for (int m = 16; m >= 1; m >>= 1) s += __shfl_xor(s, m, 32);
  if (lane == 0) kinv[row] = 1.0f / fmaxf(sqrtf(s), 1e-12f);
}

// ---------------- fused bf16-WMMA GEMM + running top-8 ----------------
// grid = (B_Q/MTILE, NCHUNK): x-fastest so the 16 WGs sharing a key chunk are
// adjacent in launch order and the 8MB chunk is served from L2, not HBM.
// block = 128 (4 waves, 2x2 wave grid, 32x64 output per wave)
__global__ __launch_bounds__(128) void gemm_topk(const float* __restrict__ keys,
                                                 const __bf16* __restrict__ qn,
                                                 const float* __restrict__ kinv,
                                                 float* __restrict__ cand_s,
                                                 int* __restrict__ cand_i) {
  __shared__ __bf16 As[MTILE][KC];        // 8 KB
  __shared__ __bf16 Bs[NBLK][KC];         // 16 KB
  __shared__ float  Sc[MTILE][NBLK + 1];  // 33 KB score staging

  const int rb    = blockIdx.x;
  const int chunk = blockIdx.y;
  const int t     = threadIdx.x;
  const int lane  = t & 31;
  const int wave  = t >> 5;
  const int wy    = wave >> 1, wx = wave & 1;
  const int half  = lane >> 4, lan16 = lane & 15;

  // running per-row top-8 (thread t owns query row t, t < MTILE)
  float ts[KTOP];
  int   ti[KTOP];
#pragma unroll
  for (int j = 0; j < KTOP; ++j) { ts[j] = NEG_INF; ti[j] = 0; }

  for (int nb = 0; nb < CHUNK / NBLK; ++nb) {
    const int key0 = chunk * CHUNK + nb * NBLK;
    if (nb + 1 < CHUNK / NBLK)
      __builtin_prefetch(keys + (size_t)(key0 + NBLK) * D_DIM, 0, 0);

    v8f acc[2][4] = {};

    for (int kc = 0; kc < D_DIM / KC; ++kc) {
      __syncthreads(); // previous compute done before restaging As/Bs
#ifdef HAVE_TDM
      if (t < 32) {
        // Tensor DMA: 64x64 bf16 tile of the 1024x512 qn tensor -> As
        unsigned long long gaddr = (unsigned long long)(const void*)qn +
                                   (unsigned long long)(rb * MTILE) * (D_DIM * 2) +
                                   (unsigned long long)kc * (KC * 2);
        unsigned lds_addr = (unsigned)(size_t)(void*)&As[0][0];
        v4u g0;
        g0[0] = 1u;                                            // count=1
        g0[1] = lds_addr;                                      // lds_addr
        g0[2] = (unsigned)(gaddr & 0xFFFFFFFFull);             // global_addr lo
        g0[3] = (unsigned)((gaddr >> 32) & 0x1FFFFFFull) |     // global_addr hi
                (2u << 30);                                    // type=2 (image)
        v8i g1;
        g1[0] = 0x10000;            // workgroup_mask=0, data_size=1 (2B)
        g1[1] = (D_DIM & 0xFFFF) << 16;                        // tensor_dim0 lo16
        g1[2] = ((D_DIM >> 16) & 0xFFFF) | ((B_Q & 0xFFFF) << 16); // dim0 hi / dim1 lo
        g1[3] = ((B_Q >> 16) & 0xFFFF) | (KC << 16);           // dim1 hi / tile_dim0
        g1[4] = MTILE;                                         // tile_dim1, tile_dim2=0
        g1[5] = D_DIM;                                         // tensor_dim0_stride lo32
        g1[6] = 0;                                             // stride0 hi / stride1 lo
        g1[7] = 0;                                             // stride1 hi
        v4i g2 = {0, 0, 0, 0};
        v4i g3 = {0, 0, 0, 0};
#if __clang_major__ >= 23
        v8i g4 = {0, 0, 0, 0, 0, 0, 0, 0};
        __builtin_amdgcn_tensor_load_to_lds(g0, g1, g2, g3, g4, 0);
#else
        __builtin_amdgcn_tensor_load_to_lds(g0, g1, g2, g3, 0);
#endif
      }
#else
      {
        // stage A: 64x64 bf16 queries (already normalized)
        int r = t >> 1, h = t & 1;
        const uint4* src =
            (const uint4*)(qn + (size_t)(rb * MTILE + r) * D_DIM + kc * KC + h * 32);
        uint4* dst = (uint4*)(&As[r][h * 32]);
#pragma unroll
        for (int i = 0; i < 4; ++i) dst[i] = src[i];
      }
#endif
      // stage B: 128x64 fp32 keys -> scaled bf16 (two half-row passes)
#pragma unroll
      for (int p = 0; p < 2; ++p) {
        int r = (t >> 1) + p * 64;
        int h = t & 1;
        int key = key0 + r;
        float sc = kinv[key];
        const float4* src =
            (const float4*)(keys + (size_t)key * D_DIM + kc * KC + h * 32);
        uint2* dst = (uint2*)(&Bs[r][h * 32]);
#pragma unroll
        for (int i = 0; i < 8; ++i) {
          float4 v = src[i];
          dst[i] = make_uint2(pk_bf16(v.x * sc, v.y * sc),
                              pk_bf16(v.z * sc, v.w * sc));
        }
      }
#ifdef HAVE_TDM
      if (t < 32) __builtin_amdgcn_s_wait_tensorcnt(0);
#endif
      __syncthreads();

#pragma unroll
      for (int ks = 0; ks < 2; ++ks) {
        const int kk = ks * 32;
        v16bf afrag[2], bfrag[4];
#pragma unroll
        for (int i = 0; i < 2; ++i) {
          int row = wy * 32 + i * 16 + lan16;
          v8bf lo = *(const v8bf*)&As[row][kk + 8 * half];
          v8bf hi = *(const v8bf*)&As[row][kk + 16 + 8 * half];
#pragma unroll
          for (int e = 0; e < 8; ++e) { afrag[i][e] = lo[e]; afrag[i][e + 8] = hi[e]; }
        }
#pragma unroll
        for (int j = 0; j < 4; ++j) {
          int col = wx * 64 + j * 16 + lan16;
          v8bf blo = *(const v8bf*)&Bs[col][kk + 8 * half];
          v8bf bhi = *(const v8bf*)&Bs[col][kk + 16 + 8 * half];
#pragma unroll
          for (int e = 0; e < 8; ++e) { bfrag[j][e] = blo[e]; bfrag[j][e + 8] = bhi[e]; }
        }
#pragma unroll
        for (int i = 0; i < 2; ++i)
#pragma unroll
          for (int j = 0; j < 4; ++j)
            acc[i][j] = __builtin_amdgcn_wmma_f32_16x16x32_bf16(
                false, afrag[i], false, bfrag[j], (short)0, acc[i][j], false, false);
      }
    }

    // stage 64x128 fp32 scores to LDS (wait for previous scan readers)
    __syncthreads();
#pragma unroll
    for (int i = 0; i < 2; ++i)
#pragma unroll
      for (int j = 0; j < 4; ++j)
#pragma unroll
        for (int r = 0; r < 8; ++r)
          Sc[wy * 32 + i * 16 + half * 8 + r][wx * 64 + j * 16 + lan16] = acc[i][j][r];
    __syncthreads();

    // running top-8 insertion, one thread per query row
    if (t < MTILE) {
      for (int c = 0; c < NBLK; ++c) {
        float s = Sc[t][c];
        if (s > ts[KTOP - 1]) {
          int p = KTOP - 1;
          while (p > 0 && s > ts[p - 1]) { ts[p] = ts[p - 1]; ti[p] = ti[p - 1]; --p; }
          ts[p] = s; ti[p] = key0 + c;
        }
      }
    }
  }

  if (t < MTILE) {
    int q = rb * MTILE + t;
    size_t base = ((size_t)q * NCHUNK + chunk) * KTOP;
#pragma unroll
    for (int j = 0; j < KTOP; ++j) { cand_s[base + j] = ts[j]; cand_i[base + j] = ti[j]; }
  }
}

// ---------------- final merge of 512 candidates + value gather ----------------
__global__ __launch_bounds__(128) void merge_gather(const float* __restrict__ values,
                                                    const float* __restrict__ cand_s,
                                                    const int* __restrict__ cand_i,
                                                    float* __restrict__ out_vals,
                                                    float* __restrict__ out_scores,
                                                    int* __restrict__ out_idx) {
  __shared__ float ss[NCHUNK * KTOP];
  __shared__ int   ii[NCHUNK * KTOP];
  __shared__ float rs[128];
  __shared__ int   rp[128];
  __shared__ float best_s[KTOP];
  __shared__ int   best_i[KTOP];

  const int q = blockIdx.x;
  const int tid = threadIdx.x;
  for (int c = tid; c < NCHUNK * KTOP; c += 128) {
    ss[c] = cand_s[(size_t)q * NCHUNK * KTOP + c];
    ii[c] = cand_i[(size_t)q * NCHUNK * KTOP + c];
  }
  __syncthreads();

  for (int r = 0; r < KTOP; ++r) {
    float m = NEG_INF; int mp = 0;
    for (int c = tid; c < NCHUNK * KTOP; c += 128)
      if (ss[c] > m) { m = ss[c]; mp = c; }
    rs[tid] = m; rp[tid] = mp;
    __syncthreads();
#pragma unroll
    for (int off = 64; off > 0; off >>= 1) {
      if (tid < off && rs[tid + off] > rs[tid]) { rs[tid] = rs[tid + off]; rp[tid] = rp[tid + off]; }
      __syncthreads();
    }
    if (tid == 0) { best_s[r] = rs[0]; best_i[r] = ii[rp[0]]; ss[rp[0]] = NEG_INF; }
    __syncthreads();
  }

  if (tid < KTOP) {
    out_scores[(size_t)q * KTOP + tid] = best_s[tid];
    out_idx[(size_t)q * KTOP + tid] = best_i[tid];
  }
#pragma unroll
  for (int r = 0; r < KTOP; ++r) {
    const float4* src = (const float4*)(values + (size_t)best_i[r] * V_DIM);
    float4* dst = (float4*)(out_vals + ((size_t)q * KTOP + r) * V_DIM);
    dst[tid] = src[tid]; // 128 threads x 16B = 512 floats
  }
}

extern "C" void kernel_launch(void* const* d_in, const int* in_sizes, int n_in,
                              void* d_out, int out_size, void* d_ws, size_t ws_size,
                              hipStream_t stream) {
  const float* query  = (const float*)d_in[0];
  const float* keys   = (const float*)d_in[1];
  const float* values = (const float*)d_in[2];
  // d_in[3] is k == 8 (compile-time constant here)

  char* ws = (char*)d_ws;
  __bf16* qn   = (__bf16*)ws;                                   // 1 MB
  float*  kinv = (float*)(ws + (size_t)B_Q * D_DIM * 2);        // 1 MB
  float*  cand_s = (float*)((char*)kinv + (size_t)C_K * 4);     // 2 MB
  int*    cand_i = (int*)((char*)cand_s + (size_t)B_Q * NCHUNK * KTOP * 4); // 2 MB

  float* out_vals   = (float*)d_out;
  float* out_scores = out_vals + (size_t)B_Q * KTOP * V_DIM;
  int*   out_idx    = (int*)(out_scores + (size_t)B_Q * KTOP);

  prep_query<<<B_Q / 8, 256, 0, stream>>>(query, qn);
  prep_keys<<<C_K / 8, 256, 0, stream>>>(keys, kinv);
  gemm_topk<<<dim3(B_Q / MTILE, NCHUNK), 128, 0, stream>>>(keys, qn, kinv, cand_s, cand_i);
  merge_gather<<<B_Q, 128, 0, stream>>>(values, cand_s, cand_i, out_vals, out_scores, out_idx);
}
